// MessagePassing_22204980920652
// MI455X (gfx1250) — matrix operand order repaired and checked by reading, not compiled, not run on hardware
//
#include <hip/hip_runtime.h>

// ---------------------------------------------------------------------------
// CDNA5 / gfx1250 hypergraph message-passing. All matmuls via
// v_wmma_f32_16x16x32_bf16 (bf16 operands, f32 accumulate), float4 global
// staging -> packed bf16 LDS tiles -> ds_load_b128 fragment gathers.
// ---------------------------------------------------------------------------

typedef __bf16 bf16;
typedef bf16  v16bf __attribute__((ext_vector_type(16)));
typedef bf16  bf16x4 __attribute__((ext_vector_type(4)));
typedef float v8f   __attribute__((ext_vector_type(8)));

#define NB   8      // batch
#define NM   1024   // nodes (seq len)
#define NE   1536   // edges
#define ND   256    // model dim
#define NH   8      // heads
#define NDH  32     // head dim
#define QKVD 768    // 3*ND
#define LDT  40     // LDS tile row stride (bf16) -> 80B, conflict-free b128 gathers

__device__ __forceinline__ v8f wmma_bf16(v16bf a, v16bf b, v8f c) {
  return __builtin_amdgcn_wmma_f32_16x16x32_bf16(
      false, a, false, b, (short)0, c, false, false);
}

// A fragment: 16x32 bf16. lane(0..15)->M; VGPR v holds
// K = ((v>>2)<<4) + (half<<3) + ((v&3)<<1) (+1 in hi bf16).
__device__ __forceinline__ void load_a_frag(const bf16* As, int rowbase,
                                            int lane, v16bf& a) {
  const int r  = rowbase + (lane & 15);
  const int hf = lane >> 4;
#pragma unroll
  for (int v = 0; v < 8; ++v) {
    const int k = ((v >> 2) << 4) + (hf << 3) + ((v & 3) << 1);
    a[2 * v]     = As[r * LDT + k];
    a[2 * v + 1] = As[r * LDT + k + 1];
  }
}

// B fragment from transposed [N][K] LDS tile. lane(0..15)->N;
// VGPR v holds K = ((v>>2)<<4) + (half<<3) + ((v&3)<<1) (+1 in hi bf16).
__device__ __forceinline__ void load_b_frag(const bf16* Bt, int colbase,
                                            int lane, v16bf& b) {
  const int n  = colbase + (lane & 15);
  const int hf = lane >> 4;
#pragma unroll
  for (int v = 0; v < 8; ++v) {
    const int k = ((v >> 2) << 4) + (hf << 3) + ((v & 3) << 1);
    b[2 * v]     = Bt[n * LDT + k];
    b[2 * v + 1] = Bt[n * LDT + k + 1];
  }
}

__device__ __forceinline__ void cvt_store4(bf16* dst, float4 f) {
  bf16x4 h;
  h[0] = (bf16)f.x; h[1] = (bf16)f.y; h[2] = (bf16)f.z; h[3] = (bf16)f.w;
  *(bf16x4*)dst = h;   // ds_store_b64
}

// ---------------------------------------------------------------------------
// GEMM: C[M,N] = A[M,K] @ W[N,K]^T (+bias). W row-major == B^T (K-contiguous).
// 256 threads = 8 waves; block tile 64x64; wave tile 16x32; K step 32.
// ---------------------------------------------------------------------------
__global__ __launch_bounds__(256) void gemm_a_wt(
    const float* __restrict__ A, const float* __restrict__ W,
    const float* __restrict__ bias, float* __restrict__ C,
    int M, int N, int K) {
  __shared__ bf16 As[64 * LDT];
  __shared__ bf16 Bs[64 * LDT];
  const int tid = threadIdx.x, lane = tid & 31, wave = tid >> 5;
  const int mbase = blockIdx.x * 64, nbase = blockIdx.y * 64;
  const int wm = (wave >> 1) * 16, wn = (wave & 1) * 32;
  v8f acc0 = {}, acc1 = {};
  for (int kb = 0; kb < K; kb += 32) {
    float4 ar[2], br[2];
#pragma unroll
    for (int v = 0; v < 2; ++v) {  // A: 64 rows x 8 float4
      const int lin = v * 256 + tid, q = lin & 7, mr = lin >> 3;
      ar[v] = *(const float4*)&A[(size_t)(mbase + mr) * K + kb + 4 * q];
      br[v] = *(const float4*)&W[(size_t)(nbase + mr) * K + kb + 4 * q];
    }
    if (kb + 32 < K)
      __builtin_prefetch(&A[(size_t)(mbase + (tid >> 2)) * K + kb + 32], 0, 1);
    __syncthreads();  // previous frag reads done
#pragma unroll
    for (int v = 0; v < 2; ++v) {
      const int lin = v * 256 + tid, q = lin & 7, mr = lin >> 3;
      cvt_store4(&As[mr * LDT + 4 * q], ar[v]);
      cvt_store4(&Bs[mr * LDT + 4 * q], br[v]);
    }
    __syncthreads();
    v16bf a, b0, b1;
    load_a_frag(As, wm, lane, a);
    load_b_frag(Bs, wn, lane, b0);
    load_b_frag(Bs, wn + 16, lane, b1);
    acc0 = wmma_bf16(a, b0, acc0);
    acc1 = wmma_bf16(a, b1, acc1);
  }
  const int l15 = lane & 15, hf = lane >> 4;
  const int n0 = nbase + wn + l15, n1 = n0 + 16;
  const float bv0 = bias ? bias[n0] : 0.0f;
  const float bv1 = bias ? bias[n1] : 0.0f;
#pragma unroll
  for (int i = 0; i < 8; ++i) {
    const size_t m = (size_t)(mbase + wm + i + 8 * hf);
    C[m * N + n0] = acc0[i] + bv0;
    C[m * N + n1] = acc1[i] + bv1;
  }
}

// ---------------------------------------------------------------------------
// scores[b] = inc[b]^T @ attn[b] : [E,M] @ [M,D].  A gathered transposed
// (float4 along e, scalar transpose stores); B transposed-staged.
// ---------------------------------------------------------------------------
__global__ __launch_bounds__(256) void gemm_incT_attn(
    const float* __restrict__ inc, const float* __restrict__ attn,
    float* __restrict__ C) {
  __shared__ bf16 As[64 * LDT];
  __shared__ bf16 Bs[64 * LDT];
  const int tid = threadIdx.x, lane = tid & 31, wave = tid >> 5;
  const int b = blockIdx.z;
  const int ebase = blockIdx.x * 64, nbase = blockIdx.y * 64;
  const int wm = (wave >> 1) * 16, wn = (wave & 1) * 32;
  v8f acc0 = {}, acc1 = {};
  for (int kb = 0; kb < NM; kb += 32) {
    float4 ar[2], br[2];
#pragma unroll
    for (int v = 0; v < 2; ++v) {
      const int lin = v * 256 + tid, eq = lin & 15, kc = lin >> 4;
      ar[v] = *(const float4*)&inc[((size_t)b * NM + kb + kc) * NE + ebase + 4 * eq];
      br[v] = *(const float4*)&attn[((size_t)b * NM + kb + kc) * ND + nbase + 4 * eq];
    }
    __syncthreads();
#pragma unroll
    for (int v = 0; v < 2; ++v) {
      const int lin = v * 256 + tid, eq = lin & 15, kc = lin >> 4;
      const float ta[4] = {ar[v].x, ar[v].y, ar[v].z, ar[v].w};
      const float tb[4] = {br[v].x, br[v].y, br[v].z, br[v].w};
#pragma unroll
      for (int j = 0; j < 4; ++j) {
        As[(4 * eq + j) * LDT + kc] = (bf16)ta[j];
        Bs[(4 * eq + j) * LDT + kc] = (bf16)tb[j];
      }
    }
    __syncthreads();
    v16bf a, b0, b1;
    load_a_frag(As, wm, lane, a);
    load_b_frag(Bs, wn, lane, b0);
    load_b_frag(Bs, wn + 16, lane, b1);
    acc0 = wmma_bf16(a, b0, acc0);
    acc1 = wmma_bf16(a, b1, acc1);
  }
  const int l15 = lane & 15, hf = lane >> 4;
  const int n0 = nbase + wn + l15, n1 = n0 + 16;
#pragma unroll
  for (int i = 0; i < 8; ++i) {
    const size_t e = (size_t)b * NE + ebase + wm + i + 8 * hf;
    C[e * ND + n0] = acc0[i];
    C[e * ND + n1] = acc1[i];
  }
}

// ---------------------------------------------------------------------------
// wef[b] = inc[b] @ edge[b] : [M,E] @ [E,D]
// ---------------------------------------------------------------------------
__global__ __launch_bounds__(256) void gemm_wef(
    const float* __restrict__ inc, const float* __restrict__ edge,
    float* __restrict__ C) {
  __shared__ bf16 As[64 * LDT];
  __shared__ bf16 Bs[64 * LDT];
  const int tid = threadIdx.x, lane = tid & 31, wave = tid >> 5;
  const int b = blockIdx.z;
  const int mbase = blockIdx.x * 64, nbase = blockIdx.y * 64;
  const int wm = (wave >> 1) * 16, wn = (wave & 1) * 32;
  v8f acc0 = {}, acc1 = {};
  for (int kb = 0; kb < NE; kb += 32) {
    float4 ar[2], br[2];
#pragma unroll
    for (int v = 0; v < 2; ++v) {
      const int lin = v * 256 + tid;
      const int q = lin & 7, mr = lin >> 3;          // A: K-contiguous
      ar[v] = *(const float4*)&inc[((size_t)b * NM + mbase + mr) * NE + kb + 4 * q];
      const int nq = lin & 15, kc = lin >> 4;        // B: N-contiguous (transpose)
      br[v] = *(const float4*)&edge[((size_t)b * NE + kb + kc) * ND + nbase + 4 * nq];
    }
    __syncthreads();
#pragma unroll
    for (int v = 0; v < 2; ++v) {
      const int lin = v * 256 + tid;
      const int q = lin & 7, mr = lin >> 3;
      cvt_store4(&As[mr * LDT + 4 * q], ar[v]);
      const int nq = lin & 15, kc = lin >> 4;
      const float tb[4] = {br[v].x, br[v].y, br[v].z, br[v].w};
#pragma unroll
      for (int j = 0; j < 4; ++j)
        Bs[(4 * nq + j) * LDT + kc] = (bf16)tb[j];
    }
    __syncthreads();
    v16bf a, b0, b1;
    load_a_frag(As, wm, lane, a);
    load_b_frag(Bs, wn, lane, b0);
    load_b_frag(Bs, wn + 16, lane, b1);
    acc0 = wmma_bf16(a, b0, acc0);
    acc1 = wmma_bf16(a, b1, acc1);
  }
  const int l15 = lane & 15, hf = lane >> 4;
  const int n0 = nbase + wn + l15, n1 = n0 + 16;
#pragma unroll
  for (int i = 0; i < 8; ++i) {
    const size_t m = (size_t)b * NM + mbase + wm + i + 8 * hf;
    C[m * ND + n0] = acc0[i];
    C[m * ND + n1] = acc1[i];
  }
}

// ---------------------------------------------------------------------------
// Flash attention over qkv[b,s,3d]. 128 threads = 4 waves; each wave owns 16
// q rows; the block shares (b,h) so K/V chunks are staged cooperatively.
// dh=32 == WMMA K. Online softmax; P routed C->A layout through LDS.
// ---------------------------------------------------------------------------
__global__ __launch_bounds__(128) void flash_attn(
    const float* __restrict__ qkv, float* __restrict__ O) {
  __shared__ bf16 P[4 * 16 * LDT];     // per-wave probabilities, A layout
  __shared__ bf16 Ks[32 * LDT];        // [key][dh]  (dh contiguous)
  __shared__ bf16 Vt[32 * LDT];        // [dh][key]  (key contiguous)
  const int tid = threadIdx.x;
  const int wave = tid >> 5, lane = tid & 31;
  const int l15 = lane & 15, hf = lane >> 4;
  const int bid = blockIdx.x;          // NB*NH*(NM/64)
  const int qt = bid & 15, h = (bid >> 4) & 7, b = bid >> 7;
  const int qbase = qt * 64 + wave * 16;
  const float scale = 0.17677669529663687f;  // 1/sqrt(32)

  v16bf qa;
  {
    const float* qrow = qkv + ((size_t)(b * NM + qbase + l15) * QKVD + h * NDH);
#pragma unroll
    for (int v = 0; v < 8; ++v) {
      const int k = ((v >> 2) << 4) + (hf << 3) + ((v & 3) << 1);
      qa[2 * v] = (bf16)qrow[k];
      qa[2 * v + 1] = (bf16)qrow[k + 1];
    }
  }
  float mrow[8], lrow[8];
#pragma unroll
  for (int i = 0; i < 8; ++i) { mrow[i] = -3.0e38f; lrow[i] = 0.0f; }
  v8f o0 = {}, o1 = {};

  for (int kb = 0; kb < NM; kb += 32) {
    float4 kreg[2], vreg[2];
#pragma unroll
    for (int v = 0; v < 2; ++v) {      // 32 keys x 8 float4 of dh
      const int lin = v * 128 + tid, q = lin & 7, key = lin >> 3;
      const size_t base = (size_t)(b * NM + kb + key) * QKVD + h * NDH + 4 * q;
      kreg[v] = *(const float4*)&qkv[base + ND];
      vreg[v] = *(const float4*)&qkv[base + 2 * ND];
    }
    __syncthreads();                   // previous-iteration reads done
#pragma unroll
    for (int v = 0; v < 2; ++v) {
      const int lin = v * 128 + tid, q = lin & 7, key = lin >> 3;
      cvt_store4(&Ks[key * LDT + 4 * q], kreg[v]);
      const float tv[4] = {vreg[v].x, vreg[v].y, vreg[v].z, vreg[v].w};
#pragma unroll
      for (int j = 0; j < 4; ++j)
        Vt[(4 * q + j) * LDT + key] = (bf16)tv[j];   // transpose to [dh][key]
    }
    __syncthreads();

    v16bf kf0, kf1;
    load_b_frag(Ks, 0, lane, kf0);     // keys kb..kb+15
    load_b_frag(Ks, 16, lane, kf1);    // keys kb+16..kb+31
    v8f z = {};
    v8f s0 = wmma_bf16(qa, kf0, z);
    v8f s1 = wmma_bf16(qa, kf1, z);

    bf16* Pw = &P[wave * 16 * LDT];
#pragma unroll
    for (int i = 0; i < 8; ++i) {
      const float si0 = s0[i] * scale, si1 = s1[i] * scale;
      float t = fmaxf(si0, si1);
#pragma unroll
      for (int m = 1; m < 16; m <<= 1) t = fmaxf(t, __shfl_xor(t, m, 16));
      const float mnew = fmaxf(mrow[i], t);
      const float f = __expf(mrow[i] - mnew);
      const float p0 = __expf(si0 - mnew), p1 = __expf(si1 - mnew);
      float rs = p0 + p1;
#pragma unroll
      for (int m = 1; m < 16; m <<= 1) rs += __shfl_xor(rs, m, 16);
      lrow[i] = lrow[i] * f + rs;
      mrow[i] = mnew;
      o0[i] *= f; o1[i] *= f;
      const int row = i + 8 * hf;
      Pw[row * LDT + l15] = (bf16)p0;
      Pw[row * LDT + 16 + l15] = (bf16)p1;
    }
    // same-wave P write->read: DScnt ordering handled by the compiler
    v16bf pa, vf0, vf1;
    load_a_frag(Pw, 0, lane, pa);
    load_b_frag(Vt, 0, lane, vf0);     // dh 0..15
    load_b_frag(Vt, 16, lane, vf1);    // dh 16..31
    o0 = wmma_bf16(pa, vf0, o0);
    o1 = wmma_bf16(pa, vf1, o1);
  }
#pragma unroll
  for (int i = 0; i < 8; ++i) {
    const float inv = 1.0f / lrow[i];
    float* orow = O + ((size_t)(b * NM + qbase + i + 8 * hf) * ND + h * NDH);
    orow[l15] = o0[i] * inv;
    orow[16 + l15] = o1[i] * inv;
  }
}

// ---------------------------------------------------------------------------
// Column softmax over the edge axis (axis=1) fused with scores*w, in place.
// ---------------------------------------------------------------------------
__global__ __launch_bounds__(256) void col_softmax_mul(float* __restrict__ S) {
  __shared__ float red[4][64];
  __shared__ float cmaxs[64], csums[64];
  const int b = blockIdx.x >> 2, slice = blockIdx.x & 3;
  const int c = threadIdx.x & 63, seg = threadIdx.x >> 6;
  float* base = S + (size_t)b * NE * ND + slice * 64 + c;
  float lmax = -3.0e38f;
  for (int r = seg; r < NE; r += 4) lmax = fmaxf(lmax, base[(size_t)r * ND]);
  red[seg][c] = lmax;
  __syncthreads();
  if (seg == 0)
    cmaxs[c] = fmaxf(fmaxf(red[0][c], red[1][c]), fmaxf(red[2][c], red[3][c]));
  __syncthreads();
  const float cm = cmaxs[c];
  float lsum = 0.0f;
  for (int r = seg; r < NE; r += 4) lsum += __expf(base[(size_t)r * ND] - cm);
  red[seg][c] = lsum;
  __syncthreads();
  if (seg == 0) csums[c] = red[0][c] + red[1][c] + red[2][c] + red[3][c];
  __syncthreads();
  const float inv = 1.0f / csums[c];
  for (int r = seg; r < NE; r += 4) {
    const float v = base[(size_t)r * ND];
    base[(size_t)r * ND] = v * (__expf(v - cm) * inv);
  }
}

// ---------------------------------------------------------------------------
// LayerNorm over d=256 (+ optional residual mix: a*prev + (1-a)*ln + prev).
// ---------------------------------------------------------------------------
__global__ __launch_bounds__(256) void layernorm_mix(
    const float* __restrict__ X, const float* __restrict__ g,
    const float* __restrict__ bt, const float* prev, const float* alpha_p,
    float* __restrict__ Y, int R) {
  const int wave = threadIdx.x >> 5, lane = threadIdx.x & 31;
  const int row = blockIdx.x * 8 + wave;
  const float* x = X + (size_t)row * ND;
  float v[8], s = 0.0f;
#pragma unroll
  for (int j = 0; j < 8; ++j) { v[j] = x[lane + 32 * j]; s += v[j]; }
#pragma unroll
  for (int m = 1; m < 32; m <<= 1) s += __shfl_xor(s, m, 32);
  const float mu = s * (1.0f / ND);
  float var = 0.0f;
#pragma unroll
  for (int j = 0; j < 8; ++j) { const float d = v[j] - mu; var += d * d; }
#pragma unroll
  for (int m = 1; m < 32; m <<= 1) var += __shfl_xor(var, m, 32);
  const float rstd = rsqrtf(var * (1.0f / ND) + 1e-5f);
  const float a = prev ? *alpha_p : 0.0f;
#pragma unroll
  for (int j = 0; j < 8; ++j) {
    const int cc = lane + 32 * j;
    float o = (v[j] - mu) * rstd * g[cc] + bt[cc];
    if (prev) {
      const float p = prev[(size_t)row * ND + cc];
      o = a * p + (1.0f - a) * o + p;
    }
    Y[(size_t)row * ND + cc] = o;
  }
}

// edge_{l+1} = a*edge_l + (1-a)*base + edge_l = (1+a)*edge_l + (1-a)*base
__global__ void edge_recur(const float* eold, const float* base,
                           const float* __restrict__ alpha_p,
                           float* __restrict__ out, size_t n) {
  const size_t i = (size_t)blockIdx.x * blockDim.x + threadIdx.x;
  if (i < n) {
    const float a = *alpha_p;
    out[i] = (1.0f + a) * eold[i] + (1.0f - a) * base[i];
  }
}

// ---------------------------------------------------------------------------
extern "C" void kernel_launch(void* const* d_in, const int* in_sizes, int n_in,
                              void* d_out, int out_size, void* d_ws, size_t ws_size,
                              hipStream_t stream) {
  const float* features = (const float*)d_in[0];
  const float* inc      = (const float*)d_in[1];
  const float* vc_Win   = (const float*)d_in[2];
  const float* vc_bin   = (const float*)d_in[3];
  const float* vc_Wout  = (const float*)d_in[4];
  const float* vc_bout  = (const float*)d_in[5];
  const float* vc_Wproj = (const float*)d_in[6];
  const float* vc_ln_g  = (const float*)d_in[7];
  const float* vc_ln_b  = (const float*)d_in[8];
  const float* vc_alpha = (const float*)d_in[9];
  // d_in[10..13]: ec MHA weights -> dead code in the reference, skipped
  const float* ec_Wproj = (const float*)d_in[14];
  const float* ec_ln_g  = (const float*)d_in[15];
  const float* ec_ln_b  = (const float*)d_in[16];
  const float* ec_alpha = (const float*)d_in[17];

  const size_t MD = (size_t)NB * NM * ND;
  const size_t ED = (size_t)NB * NE * ND;
  const size_t ME = (size_t)NB * NM * NE;
  const size_t QK = (size_t)NB * NM * QKVD;

  float* ws    = (float*)d_ws;
  float* qkv   = ws;
  float* o     = qkv + QK;
  float* attn  = o + MD;
  float* sc    = attn + MD;
  float* ebase = sc + ED;
  float* e1    = ebase + ED;
  float* tmpE  = e1 + ED;
  float* wef   = tmpE + ED;
  float* tmpM  = wef + MD;
  float* node0 = tmpM + MD;
  float* node1 = node0 + MD;

  float* out_node = (float*)d_out;
  float* out_edge = out_node + MD;
  float* out_inc  = out_edge + ED;

  const dim3 blk(256);

  // ---- layer-invariant base edge path (features never change) -------------
  gemm_a_wt<<<dim3(NB * NM / 64, QKVD / 64), blk, 0, stream>>>(
      features, vc_Win, vc_bin, qkv, NB * NM, QKVD, ND);
  flash_attn<<<dim3(NB * NH * (NM / 64)), dim3(128), 0, stream>>>(qkv, o);
  gemm_a_wt<<<dim3(NB * NM / 64, ND / 64), blk, 0, stream>>>(
      o, vc_Wout, vc_bout, attn, NB * NM, ND, ND);
  gemm_incT_attn<<<dim3(NE / 64, ND / 64, NB), blk, 0, stream>>>(inc, attn, sc);
  col_softmax_mul<<<dim3(NB * 4), blk, 0, stream>>>(sc);
  gemm_a_wt<<<dim3(NB * NE / 64, ND / 64), blk, 0, stream>>>(
      sc, vc_Wproj, nullptr, tmpE, NB * NE, ND, ND);
  layernorm_mix<<<dim3(NB * NE / 8), blk, 0, stream>>>(
      tmpE, vc_ln_g, vc_ln_b, nullptr, nullptr, ebase, NB * NE);

  // ---- layer 0: node_0 = edge_conv(inc, edge_0=ebase, prev=features) ------
  gemm_wef<<<dim3(NM / 64, ND / 64, NB), blk, 0, stream>>>(inc, ebase, wef);
  gemm_a_wt<<<dim3(NB * NM / 64, ND / 64), blk, 0, stream>>>(
      wef, ec_Wproj, nullptr, tmpM, NB * NM, ND, ND);
  layernorm_mix<<<dim3(NB * NM / 8), blk, 0, stream>>>(
      tmpM, ec_ln_g, ec_ln_b, features, ec_alpha, node0, NB * NM);

  // ---- layer 1 -------------------------------------------------------------
  edge_recur<<<dim3((unsigned)((ED + 255) / 256)), blk, 0, stream>>>(
      ebase, ebase, vc_alpha, e1, ED);
  gemm_wef<<<dim3(NM / 64, ND / 64, NB), blk, 0, stream>>>(inc, e1, wef);
  gemm_a_wt<<<dim3(NB * NM / 64, ND / 64), blk, 0, stream>>>(
      wef, ec_Wproj, nullptr, tmpM, NB * NM, ND, ND);
  layernorm_mix<<<dim3(NB * NM / 8), blk, 0, stream>>>(
      tmpM, ec_ln_g, ec_ln_b, node0, ec_alpha, node1, NB * NM);

  // ---- layer 2 (final edge/node straight to d_out) -------------------------
  edge_recur<<<dim3((unsigned)((ED + 255) / 256)), blk, 0, stream>>>(
      e1, ebase, vc_alpha, out_edge, ED);
  gemm_wef<<<dim3(NM / 64, ND / 64, NB), blk, 0, stream>>>(inc, out_edge, wef);
  gemm_a_wt<<<dim3(NB * NM / 64, ND / 64), blk, 0, stream>>>(
      wef, ec_Wproj, nullptr, tmpM, NB * NM, ND, ND);
  layernorm_mix<<<dim3(NB * NM / 8), blk, 0, stream>>>(
      tmpM, ec_ln_g, ec_ln_b, node1, ec_alpha, out_node, NB * NM);

  // ---- third output: inc_mat passthrough -----------------------------------
  hipMemcpyAsync(out_inc, inc, ME * sizeof(float),
                 hipMemcpyDeviceToDevice, stream);

  (void)in_sizes; (void)n_in; (void)out_size; (void)ws_size;
}
